// HeteroGraphSAGE_19628000542879
// MI455X (gfx1250) — matrix operand order repaired and checked by reading, not compiled, not run on hardware
//
#include <hip/hip_runtime.h>

// ---------------- problem constants (from reference) ----------------
#define H    256
#define NPN  20000
#define NGN  40000
#define NDN  8000
#define PF   2816
#define GF   200
#define DF   512
#define E_PG 400000
#define E_GP 400000
#define E_PP 600000
#define E_DD 200000

typedef __attribute__((ext_vector_type(16))) _Float16 v16h;
typedef __attribute__((ext_vector_type(8)))  float    v8f;

// ---- fragment loaders (branch-free fast path; guarded tail path) ----
// A (16x32 f16): lane l -> row M=l%16; elem j -> K = kb + 16*(l/16) + j
__device__ __forceinline__ v16h load_a_full(const float* __restrict__ xrow, int kb, int half)
{
    const float4* p0 = reinterpret_cast<const float4*>(xrow + kb + half * 16);
    const float4* p1 = reinterpret_cast<const float4*>(xrow + kb + half * 16 + 8);
    float4 q0 = p0[0], q1 = p0[1], q2 = p1[0], q3 = p1[1];
    v16h a;
    a[0]=(_Float16)q0.x; a[1]=(_Float16)q0.y; a[2]=(_Float16)q0.z; a[3]=(_Float16)q0.w;
    a[4]=(_Float16)q1.x; a[5]=(_Float16)q1.y; a[6]=(_Float16)q1.z; a[7]=(_Float16)q1.w;
    a[8]=(_Float16)q2.x; a[9]=(_Float16)q2.y; a[10]=(_Float16)q2.z; a[11]=(_Float16)q2.w;
    a[12]=(_Float16)q3.x; a[13]=(_Float16)q3.y; a[14]=(_Float16)q3.z; a[15]=(_Float16)q3.w;
    return a;
}
__device__ __forceinline__ v16h load_a_tail(const float* __restrict__ xrow, int kb, int half, int K)
{
    v16h a;
    #pragma unroll
    for (int j = 0; j < 16; ++j) {
        int k = kb + half * 16 + j;
        a[j] = (_Float16)((k < K) ? xrow[k] : 0.0f);
    }
    return a;
}
// B (32x16 f16): lane l -> col N=l%16; elem j -> K = kb + 16*(l/16) + j
__device__ __forceinline__ v16h load_b_full(const float* __restrict__ wq, int kb)
{
    const float4* p = reinterpret_cast<const float4*>(wq + kb);
    float4 q0 = p[0], q1 = p[1], q2 = p[2], q3 = p[3];
    v16h b;
    b[0]=(_Float16)q0.x; b[1]=(_Float16)q0.y; b[2]=(_Float16)q0.z; b[3]=(_Float16)q0.w;
    b[4]=(_Float16)q1.x; b[5]=(_Float16)q1.y; b[6]=(_Float16)q1.z; b[7]=(_Float16)q1.w;
    b[8]=(_Float16)q2.x; b[9]=(_Float16)q2.y; b[10]=(_Float16)q2.z; b[11]=(_Float16)q2.w;
    b[12]=(_Float16)q3.x; b[13]=(_Float16)q3.y; b[14]=(_Float16)q3.z; b[15]=(_Float16)q3.w;
    return b;
}
__device__ __forceinline__ v16h load_b_tail(const float* __restrict__ wq, int kb, int half, int K)
{
    v16h b;
    #pragma unroll
    for (int j = 0; j < 16; ++j) {
        int k = kb + half * 16 + j;
        b[j] = (_Float16)((k < K) ? wq[kb + j] : 0.0f);
    }
    return b;
}

// =====================================================================
// GEMM: Y[N x 256] = X[N x K] @ W[256 x K]^T   (fp32 in/out, f16 WMMA)
// Block = 256 threads = 8 waves. Each wave: 32 rows x 64 cols = 8 WMMA
// tiles; 2 A fragments shared across 4 B fragments per K-step (8 WMMAs
// per 6 fragment loads). grid = (ceil(N/256), 4).
// Out-of-range A rows are CLAMPED to row N-1: garbage only lands in D
// rows m>=N which are never stored -> branch-free inner loop.
// =====================================================================
__global__ __launch_bounds__(256)
void gemm_xwT(const float* __restrict__ X, const float* __restrict__ W,
              float* __restrict__ Y, int N, int K)
{
    const int wave = threadIdx.x >> 5;
    const int lane = threadIdx.x & 31;
    const int half = lane >> 4;     // 0/1
    const int lr   = lane & 15;     // 0..15

    const int row0 = (blockIdx.x * 8 + wave) * 32;
    const int col0 = blockIdx.y * 64;
    if (row0 >= N) return;          // wave-uniform exit; live waves keep EXEC all-1s

    int ar0 = row0 + lr;       if (ar0 > N - 1) ar0 = N - 1;
    int ar1 = row0 + 16 + lr;  if (ar1 > N - 1) ar1 = N - 1;
    const float* xr0 = X + (size_t)ar0 * K;
    const float* xr1 = X + (size_t)ar1 * K;
    // per-tile W row base, pre-offset by this lane's K-half
    const float* wq0 = W + (size_t)(col0 +  0 + lr) * K + half * 16;
    const float* wq1 = W + (size_t)(col0 + 16 + lr) * K + half * 16;
    const float* wq2 = W + (size_t)(col0 + 32 + lr) * K + half * 16;
    const float* wq3 = W + (size_t)(col0 + 48 + lr) * K + half * 16;

    v8f acc[8] = {};                // [t*2 + mtile]

    const int Kfull = K & ~31;
    for (int kb = 0; kb < Kfull; kb += 32) {
        v16h a0 = load_a_full(xr0, kb, half);
        v16h a1 = load_a_full(xr1, kb, half);
        v16h b0 = load_b_full(wq0, kb);
        v16h b1 = load_b_full(wq1, kb);
        v16h b2 = load_b_full(wq2, kb);
        v16h b3 = load_b_full(wq3, kb);
        acc[0] = __builtin_amdgcn_wmma_f32_16x16x32_f16(false, a0, false, b0, (short)0, acc[0], false, false);
        acc[1] = __builtin_amdgcn_wmma_f32_16x16x32_f16(false, a1, false, b0, (short)0, acc[1], false, false);
        acc[2] = __builtin_amdgcn_wmma_f32_16x16x32_f16(false, a0, false, b1, (short)0, acc[2], false, false);
        acc[3] = __builtin_amdgcn_wmma_f32_16x16x32_f16(false, a1, false, b1, (short)0, acc[3], false, false);
        acc[4] = __builtin_amdgcn_wmma_f32_16x16x32_f16(false, a0, false, b2, (short)0, acc[4], false, false);
        acc[5] = __builtin_amdgcn_wmma_f32_16x16x32_f16(false, a1, false, b2, (short)0, acc[5], false, false);
        acc[6] = __builtin_amdgcn_wmma_f32_16x16x32_f16(false, a0, false, b3, (short)0, acc[6], false, false);
        acc[7] = __builtin_amdgcn_wmma_f32_16x16x32_f16(false, a1, false, b3, (short)0, acc[7], false, false);
    }
    if (Kfull < K) {                // single guarded tail step (only K=200 hits this)
        const int kb = Kfull;
        v16h a0 = load_a_tail(xr0, kb, half, K);
        v16h a1 = load_a_tail(xr1, kb, half, K);
        v16h b0 = load_b_tail(wq0, kb, half, K);
        v16h b1 = load_b_tail(wq1, kb, half, K);
        v16h b2 = load_b_tail(wq2, kb, half, K);
        v16h b3 = load_b_tail(wq3, kb, half, K);
        acc[0] = __builtin_amdgcn_wmma_f32_16x16x32_f16(false, a0, false, b0, (short)0, acc[0], false, false);
        acc[1] = __builtin_amdgcn_wmma_f32_16x16x32_f16(false, a1, false, b0, (short)0, acc[1], false, false);
        acc[2] = __builtin_amdgcn_wmma_f32_16x16x32_f16(false, a0, false, b1, (short)0, acc[2], false, false);
        acc[3] = __builtin_amdgcn_wmma_f32_16x16x32_f16(false, a1, false, b1, (short)0, acc[3], false, false);
        acc[4] = __builtin_amdgcn_wmma_f32_16x16x32_f16(false, a0, false, b2, (short)0, acc[4], false, false);
        acc[5] = __builtin_amdgcn_wmma_f32_16x16x32_f16(false, a1, false, b2, (short)0, acc[5], false, false);
        acc[6] = __builtin_amdgcn_wmma_f32_16x16x32_f16(false, a0, false, b3, (short)0, acc[6], false, false);
        acc[7] = __builtin_amdgcn_wmma_f32_16x16x32_f16(false, a1, false, b3, (short)0, acc[7], false, false);
    }

    // D store: lane l, vgpr r -> M = r + 8*(l/16), N = l%16
    #pragma unroll
    for (int t = 0; t < 4; ++t) {
        #pragma unroll
        for (int mt = 0; mt < 2; ++mt) {
            const int mbase = row0 + mt * 16 + 8 * half;
            #pragma unroll
            for (int r = 0; r < 8; ++r) {
                int m = mbase + r;
                if (m < N) Y[(size_t)m * H + col0 + t * 16 + lr] = acc[t * 2 + mt][r];
            }
        }
    }
}

// =====================================================================
// Scatter: wave per edge; lane accumulates 8 strided floats (fully
// coalesced 128B bursts) of the projected src row into acc[dst], plus
// an edge count for the mean.
// =====================================================================
__global__ __launch_bounds__(256)
void scatter_mean_acc(const float* __restrict__ msg, const int* __restrict__ src,
                      const int* __restrict__ dst, float* __restrict__ acc,
                      float* __restrict__ cnt, int E)
{
    int e    = (blockIdx.x * blockDim.x + threadIdx.x) >> 5;
    int lane = threadIdx.x & 31;
    if (e >= E) return;
    int s = src[e], d = dst[e];
    const float* m = msg + (size_t)s * H;
    float*       a = acc + (size_t)d * H;
    #pragma unroll
    for (int j = 0; j < 8; ++j)
        atomicAdd(a + lane + j * 32, m[lane + j * 32]);
    if (lane == 0) atomicAdd(cnt + d, 1.0f);
}

// =====================================================================
// Combine: out = acc/max(cnt,1) + b + ywr ; optional row L2-normalize.
// One 256-thread block per dst row (H == 256).
// =====================================================================
__global__ __launch_bounds__(256)
void combine_rel(const float* __restrict__ acc, const float* __restrict__ cnt,
                 const float* __restrict__ bias, const float* __restrict__ ywr,
                 float* __restrict__ out, int normalize)
{
    __shared__ float red[256];
    const int row = blockIdx.x;
    const int h   = threadIdx.x;
    const size_t i = (size_t)row * H + h;
    float c = cnt[row];
    float v = acc[i] / (c > 1.0f ? c : 1.0f) + bias[h] + ywr[i];
    if (normalize) {
        red[h] = v * v;
        __syncthreads();
        #pragma unroll
        for (int s = 128; s > 0; s >>= 1) {
            if (h < s) red[h] += red[h + s];
            __syncthreads();
        }
        v = v / fmaxf(sqrtf(red[0]), 1e-12f);
    }
    out[i] = v;
}

// =====================================================================
// Hetero-aggregate + PReLU (scalar alpha)
// =====================================================================
__global__ __launch_bounds__(256)
void agg_prelu2(const float* __restrict__ a, const float* __restrict__ b,
                const float* __restrict__ alpha, float* __restrict__ out, size_t n)
{
    size_t i = (size_t)blockIdx.x * blockDim.x + threadIdx.x;
    if (i >= n) return;
    float v = a[i] + b[i];
    out[i] = v >= 0.0f ? v : (*alpha) * v;
}
__global__ __launch_bounds__(256)
void agg_prelu1(const float* __restrict__ a, const float* __restrict__ alpha,
                float* __restrict__ out, size_t n)
{
    size_t i = (size_t)blockIdx.x * blockDim.x + threadIdx.x;
    if (i >= n) return;
    float v = a[i];
    out[i] = v >= 0.0f ? v : (*alpha) * v;
}

// =====================================================================
// Host orchestration.
// Assumed d_in layout (setup_inputs dict order; params pytree-flattened
// with sorted dict keys -> "alpha" before "layers"; rel keys sorted
// dd,gp,pg,pp; leaf keys sorted Wl,Wr,b):
//   0 x_protein  1 x_go  2 x_drug
//   3,4,5 alpha[0..2]
//   6 + 12*l + {dd:0, gp:3, pg:6, pp:9} + {Wl:0, Wr:1, b:2}
//   45 ei_pg  46 ei_gp  47 ei_pp  48 ei_dd   (each [2,E]: src then dst)
// =====================================================================
extern "C" void kernel_launch(void* const* d_in, const int* in_sizes, int n_in,
                              void* d_out, int out_size, void* d_ws, size_t ws_size,
                              hipStream_t stream)
{
    const float* xp = (const float*)d_in[0];
    const float* xg = (const float*)d_in[1];
    const float* xd = (const float*)d_in[2];
    int kp = PF, kg = GF, kd = DF;

    // ---- workspace carve-out (floats) ----
    float* ws  = (float*)d_ws;
    size_t off = 0;
    auto carve = [&](size_t n) { float* p = ws + off; off += n; return p; };
    float* msg    = carve((size_t)NGN * H);   // projected src rows (max N_src)
    float* acc    = carve((size_t)NGN * H);   // scatter accumulator (max N_dst)
    float* cnt    = carve(40960);             // per-dst edge counts
    float* rel_pg = carve((size_t)NGN * H);
    float* rel_gp = carve((size_t)NPN * H);
    float* rel_pp = carve((size_t)NPN * H);
    float* rel_dd = carve((size_t)NDN * H);
    float* ywr    = msg;  // safe alias: Wr GEMM launches after scatter consumed msg
    (void)ws_size; (void)in_sizes; (void)n_in; (void)out_size;

    // d_out doubles as the per-layer x buffers (layout == concat(xp,xg,xd))
    float* out_p = (float*)d_out;
    float* out_g = out_p + (size_t)NPN * H;
    float* out_d = out_g + (size_t)NGN * H;

    auto run_rel = [&](const float* xsrc, int ksrc, int nsrc,
                       const float* xdst, int kdst, int ndst,
                       const float* Wl, const float* Wr, const float* bias,
                       const int* ei, int E, float* out, int normalize) {
        dim3 gs((nsrc + 255) / 256, H / 64);
        gemm_xwT<<<gs, 256, 0, stream>>>(xsrc, Wl, msg, nsrc, ksrc);
        hipMemsetAsync(acc, 0, (size_t)ndst * H * sizeof(float), stream);
        hipMemsetAsync(cnt, 0, (size_t)ndst * sizeof(float), stream);
        scatter_mean_acc<<<(E + 7) / 8, 256, 0, stream>>>(msg, ei, ei + E, acc, cnt, E);
        dim3 gd((ndst + 255) / 256, H / 64);
        gemm_xwT<<<gd, 256, 0, stream>>>(xdst, Wr, ywr, ndst, kdst);
        combine_rel<<<ndst, 256, 0, stream>>>(acc, cnt, bias, ywr, out, normalize);
    };

    const int* ei_pg = (const int*)d_in[45];
    const int* ei_gp = (const int*)d_in[46];
    const int* ei_pp = (const int*)d_in[47];
    const int* ei_dd = (const int*)d_in[48];

    for (int l = 0; l < 3; ++l) {
        const int pb = 6 + 12 * l;
        const float* dd_Wl = (const float*)d_in[pb + 0];
        const float* dd_Wr = (const float*)d_in[pb + 1];
        const float* dd_b  = (const float*)d_in[pb + 2];
        const float* gp_Wl = (const float*)d_in[pb + 3];
        const float* gp_Wr = (const float*)d_in[pb + 4];
        const float* gp_b  = (const float*)d_in[pb + 5];
        const float* pg_Wl = (const float*)d_in[pb + 6];
        const float* pg_Wr = (const float*)d_in[pb + 7];
        const float* pg_b  = (const float*)d_in[pb + 8];
        const float* pp_Wl = (const float*)d_in[pb + 9];
        const float* pp_Wr = (const float*)d_in[pb + 10];
        const float* pp_b  = (const float*)d_in[pb + 11];
        const int norm = (l > 0) ? 1 : 0;

        run_rel(xp, kp, NPN, xg, kg, NGN, pg_Wl, pg_Wr, pg_b, ei_pg, E_PG, rel_pg, norm);
        run_rel(xg, kg, NGN, xp, kp, NPN, gp_Wl, gp_Wr, gp_b, ei_gp, E_GP, rel_gp, norm);
        run_rel(xp, kp, NPN, xp, kp, NPN, pp_Wl, pp_Wr, pp_b, ei_pp, E_PP, rel_pp, norm);
        run_rel(xd, kd, NDN, xd, kd, NDN, dd_Wl, dd_Wr, dd_b, ei_dd, E_DD, rel_dd, norm);

        const float* alpha = (const float*)d_in[3 + l];
        const size_t np_n = (size_t)NPN * H, ng_n = (size_t)NGN * H, nd_n = (size_t)NDN * H;
        agg_prelu2<<<(unsigned)((np_n + 255) / 256), 256, 0, stream>>>(rel_gp, rel_pp, alpha, out_p, np_n);
        agg_prelu1<<<(unsigned)((ng_n + 255) / 256), 256, 0, stream>>>(rel_pg, alpha, out_g, ng_n);
        agg_prelu1<<<(unsigned)((nd_n + 255) / 256), 256, 0, stream>>>(rel_dd, alpha, out_d, nd_n);

        xp = out_p; xg = out_g; xd = out_d;
        kp = kg = kd = H;
    }
}
